// LSTMTextProcessor_84026740179697
// MI455X (gfx1250) — compile-verified, matrix operand order checked by reading, not run on hardware
//
#include <hip/hip_runtime.h>
#include <cstdint>
#include <cstddef>

// ---------------------------------------------------------------------------
// CDNA5 (gfx1250) bidirectional 2-layer LSTM + FC.
//  - All GEMMs via v_wmma_f32_16x16x32_bf16 (f32 inputs downcast to bf16,
//    f32 accumulate).
//  - GEMM A-tiles staged in LDS by the Tensor Data Mover (tensor_load_to_lds,
//    TENSORcnt), WMMA fed from LDS; W fragments register-double-buffered.
//  - Recurrence: 1 workgroup/direction, h double-buffered in LDS, c in VGPRs,
//    Whh streamed from L2; per-step y output written by tensor_store_from_lds.
// ---------------------------------------------------------------------------

typedef __attribute__((ext_vector_type(16))) __bf16 v16bf;
typedef __attribute__((ext_vector_type(8)))  float  v8f;
typedef unsigned int u32;
typedef __attribute__((ext_vector_type(4))) u32 v4u;
typedef __attribute__((ext_vector_type(4))) int v4i;
typedef __attribute__((ext_vector_type(8))) int v8i;

#define BB 64
#define TT 512
#define EE 300
#define EP 320      // E padded to multiple of 32 (WMMA K-step)
#define HH 256
#define G4 1024     // 4*H
#define H2 512      // 2*H
#define OO 512
#define BT (BB*TT)

// Toolchain TDM builtin arity dispatch: the 6-arg form ships with the
// toolchain that also provides the gfx1250 TDM header; ROCm 7.2 uses 5 args.
#if defined(__has_include)
#if __has_include(<hip/amd_detail/amd_gfx1250_TDM.h>)
#define TDM_6ARG 1
#else
#define TDM_6ARG 0
#endif
#else
#define TDM_6ARG 0
#endif

__device__ __forceinline__ float sigf(float x) { return 1.0f / (1.0f + __expf(-x)); }

__device__ __forceinline__ v8f wmma_bf16(v16bf a, v16bf b, v8f c) {
  return __builtin_amdgcn_wmma_f32_16x16x32_bf16(false, a, false, b, (short)0, c,
                                                 false, false);
}

// ---- Tensor DMA descriptor (D#) builders: 2D tile of bf16 (data_size=2B) ----
__device__ __forceinline__ v4u tdm_g0(u32 lds_off, const void* gptr) {
  uint64_t ga = (uint64_t)gptr;
  v4u g0;
  g0[0] = 1u;                                              // count=1 (valid)
  g0[1] = lds_off;                                         // LDS byte address
  g0[2] = (u32)ga;                                         // global_addr[31:0]
  g0[3] = ((u32)(ga >> 32) & 0x01FFFFFFu) | (2u << 30);    // addr[56:32], type=2
  return g0;
}
__device__ __forceinline__ v8i tdm_g1_2d(u32 row_elems, u32 nrows, u32 row_stride) {
  v8i g1;
  g1[0] = (int)(1u << 16);                                 // data_size=1 (2 bytes)
  g1[1] = (int)((row_elems & 0xFFFFu) << 16);              // tensor_dim0 lo16
  g1[2] = (int)(((row_elems >> 16) & 0xFFFFu) | ((nrows & 0xFFFFu) << 16));
  g1[3] = (int)(((nrows >> 16) & 0xFFFFu) | ((row_elems & 0xFFFFu) << 16)); // tile_dim0
  g1[4] = (int)(nrows & 0xFFFFu);                          // tile_dim1, tile_dim2=0
  g1[5] = (int)row_stride;                                 // tensor_dim0_stride lo32
  g1[6] = 0;
  g1[7] = 0;
  return g1;
}
__device__ __forceinline__ void tdm_load_2d(u32 lds_off, const void* g,
                                            u32 row_elems, u32 nrows, u32 row_stride) {
  v4i z = {0, 0, 0, 0};
#if TDM_6ARG
  v8i z8 = {0, 0, 0, 0, 0, 0, 0, 0};
  __builtin_amdgcn_tensor_load_to_lds(tdm_g0(lds_off, g), tdm_g1_2d(row_elems, nrows, row_stride),
                                      z, z, z8, 0);
#else
  __builtin_amdgcn_tensor_load_to_lds(tdm_g0(lds_off, g), tdm_g1_2d(row_elems, nrows, row_stride),
                                      z, z, 0);
#endif
}
__device__ __forceinline__ void tdm_store_2d(u32 lds_off, void* g,
                                             u32 row_elems, u32 nrows, u32 row_stride) {
  v4i z = {0, 0, 0, 0};
#if TDM_6ARG
  v8i z8 = {0, 0, 0, 0, 0, 0, 0, 0};
  __builtin_amdgcn_tensor_store_from_lds(tdm_g0(lds_off, g), tdm_g1_2d(row_elems, nrows, row_stride),
                                         z, z, z8, 0);
#else
  __builtin_amdgcn_tensor_store_from_lds(tdm_g0(lds_off, g), tdm_g1_2d(row_elems, nrows, row_stride),
                                         z, z, 0);
#endif
}

// ---------------- embedding gather + f32->bf16 + pad to EP ----------------
__global__ void embed_kernel(const int* __restrict__ text,
                             const float* __restrict__ emb,
                             __bf16* __restrict__ xb) {
  const int row = blockIdx.x;        // b*T + t
  const int e   = threadIdx.x;       // 0..EP-1
  const int tok = text[row];
  const float v = (e < EE) ? emb[(size_t)tok * EE + e] : 0.0f;
  xb[(size_t)row * EP + e] = (__bf16)v;
}

// ---------------- weight cast f32->bf16 with K padding ----------------
__global__ void castpad_kernel(const float* __restrict__ src,
                               __bf16* __restrict__ dst,
                               int rows, int cols, int colsP) {
  const int idx = blockIdx.x * blockDim.x + threadIdx.x;
  if (idx >= rows * colsP) return;
  const int r = idx / colsP, c = idx % colsP;
  dst[idx] = (c < cols) ? (__bf16)src[(size_t)r * cols + c] : (__bf16)0.0f;
}

// ---------------- WMMA GEMM:  C[M,N] = A[M,KP] * W[N,KP]^T + bias -----------
// Block = 256 thr (8 waves), tile 64(M) x 128(N). A tile (64 x KP bf16) is
// DMA'd to LDS once per block by the TDM; K-loop feeds WMMA from LDS while
// W fragments are register-double-buffered (loads one K-step ahead).
__global__ __launch_bounds__(256)
void wmma_gemm_bias(const __bf16* __restrict__ A,
                    const __bf16* __restrict__ W,
                    const float* __restrict__ bias,
                    float* __restrict__ C,
                    int N, int KP) {
  __shared__ __align__(32) __bf16 As[64 * 512];    // 64 KB max (KP<=512)

  const int lane = threadIdx.x & 31;
  const int wave = threadIdx.x >> 5;
  const int lm = lane & 15;              // row/col within 16
  const int lk = (lane >> 4) << 4;       // K half-select (0 or 16)
  const int col0 = blockIdx.x * 128 + wave * 16;
  const int row0 = blockIdx.y * 64;

  if (wave == 0) {
    tdm_load_2d((u32)(size_t)(void*)As, A + (size_t)row0 * KP,
                (u32)KP, 64u, (u32)KP);
    __builtin_amdgcn_s_wait_tensorcnt(0);
  }
  __syncthreads();

  v8f acc0 = {}, acc1 = {}, acc2 = {}, acc3 = {};

  const __bf16* wp = W + (size_t)(col0 + lm) * KP + lk;   // B: col = lane%16
  const __bf16* ap = As + (size_t)lm * KP + lk;           // A: row = lane%16

  v16bf bcur = *(const v16bf*)wp;
  for (int k0 = 0; k0 < KP; k0 += 32) {
    v16bf bnxt = bcur;
    if (k0 + 32 < KP) bnxt = *(const v16bf*)(wp + k0 + 32);   // pipeline W
    v16bf a0 = *(const v16bf*)(ap + k0);
    v16bf a1 = *(const v16bf*)(ap + 16 * KP + k0);
    v16bf a2 = *(const v16bf*)(ap + 32 * KP + k0);
    v16bf a3 = *(const v16bf*)(ap + 48 * KP + k0);
    acc0 = wmma_bf16(a0, bcur, acc0);
    acc1 = wmma_bf16(a1, bcur, acc1);
    acc2 = wmma_bf16(a2, bcur, acc2);
    acc3 = wmma_bf16(a3, bcur, acc3);
    bcur = bnxt;
  }

  const int ccol = col0 + lm;
  const float bv = bias ? bias[ccol] : 0.0f;
  const int crow = row0 + ((lane >> 4) << 3);   // C: m = vj + 8*(lane/16)
#pragma unroll
  for (int vj = 0; vj < 8; ++vj) {
    C[(size_t)(crow + vj) * N + ccol]      = acc0[vj] + bv;
    C[(size_t)(crow + 16 + vj) * N + ccol] = acc1[vj] + bv;
    C[(size_t)(crow + 32 + vj) * N + ccol] = acc2[vj] + bv;
    C[(size_t)(crow + 48 + vj) * N + ccol] = acc3[vj] + bv;
  }
}

// ---------------- recurrent LSTM: one workgroup per direction --------------
// 1024 thr = 32 waves. Wave w: 16 hidden units (all 4 gates) x 32 batch rows.
// h double-buffered in LDS; c in VGPRs; y (layer0) written per step by a
// strided 2D tensor_store_from_lds issued by wave 0.
__global__ __launch_bounds__(1024)
void lstm_kernel(const float* __restrict__ xgF, const float* __restrict__ xgB,
                 const __bf16* __restrict__ UF, const __bf16* __restrict__ UB,
                 const float* __restrict__ bhF, const float* __restrict__ bhB,
                 __bf16* __restrict__ y,        // layer0: [BT, 2H] (or null)
                 __bf16* __restrict__ hfinal,   // layer1: [B, 2H]  (or null)
                 int writeY) {
  const int dir = blockIdx.x;
  const float*  xg = dir ? xgB : xgF;
  const __bf16* U  = dir ? UB : UF;
  const float*  bh = dir ? bhB : bhF;
  const int ycol = dir * HH;

  __shared__ __align__(32) __bf16 hsm[2][BB * HH];   // 64 KB double-buffered h

  const int tid = threadIdx.x;
  for (int i = tid; i < 2 * BB * HH; i += 1024) ((__bf16*)hsm)[i] = (__bf16)0.0f;

  const int lane = tid & 31;
  const int wave = tid >> 5;
  const int lm = lane & 15;
  const int lk = (lane >> 4) << 4;
  const int lh = (lane >> 4) << 3;
  const int j0 = (wave & 15) * 16;     // hidden-unit strip
  const int m0 = (wave >> 4) * 32;     // batch-row block
  const int n  = j0 + lm;              // this lane's hidden unit

  const float bhi = bh[n];
  const float bhf = bh[HH + n];
  const float bhg = bh[2 * HH + n];
  const float bho = bh[3 * HH + n];

  float cst[2][8];
#pragma unroll
  for (int mt = 0; mt < 2; ++mt)
#pragma unroll
    for (int vj = 0; vj < 8; ++vj) cst[mt][vj] = 0.0f;

  __syncthreads();

  const __bf16* urow = U + (size_t)(j0 + lm) * HH + lk;  // gate-0 B rows

  for (int s = 0; s < TT; ++s) {
    const int t = dir ? (TT - 1 - s) : s;
    const __bf16* hcur = hsm[s & 1];
    __bf16* hnxt = hsm[(s & 1) ^ 1];

    v8f acc[2][4];
#pragma unroll
    for (int mt = 0; mt < 2; ++mt)
#pragma unroll
      for (int g = 0; g < 4; ++g) acc[mt][g] = v8f{};

#pragma unroll
    for (int k0 = 0; k0 < HH; k0 += 32) {
      v16bf a0 = *(const v16bf*)&hcur[(m0 + lm) * HH + k0 + lk];
      v16bf a1 = *(const v16bf*)&hcur[(m0 + 16 + lm) * HH + k0 + lk];
#pragma unroll
      for (int g = 0; g < 4; ++g) {
        __builtin_prefetch(urow + (size_t)g * HH * HH + k0 + 32, 0, 1);
        v16bf bfr = *(const v16bf*)(urow + (size_t)g * HH * HH + k0);
        acc[0][g] = wmma_bf16(a0, bfr, acc[0][g]);
        acc[1][g] = wmma_bf16(a1, bfr, acc[1][g]);
      }
    }

    // TDM store issued at step s-2 read buffer 'hnxt'; make sure it drained
    // before we overwrite it (TDM ops complete in order per wave).
    if (writeY && wave == 0) __builtin_amdgcn_s_wait_tensorcnt(1);
    __syncthreads();

#pragma unroll
    for (int mt = 0; mt < 2; ++mt) {
      const int mb = m0 + mt * 16 + lh;
#pragma unroll
      for (int vj = 0; vj < 8; ++vj) {
        const int m = mb + vj;                       // batch index
        const float* xr = xg + ((size_t)m * TT + t) * G4;
        const float iv = acc[mt][0][vj] + xr[n]          + bhi;
        const float fv = acc[mt][1][vj] + xr[HH + n]     + bhf;
        const float gv = acc[mt][2][vj] + xr[2 * HH + n] + bhg;
        const float ov = acc[mt][3][vj] + xr[3 * HH + n] + bho;
        const float cn = sigf(fv) * cst[mt][vj] + sigf(iv) * tanhf(gv);
        cst[mt][vj] = cn;
        const float hv = sigf(ov) * tanhf(cn);
        hnxt[m * HH + n] = (__bf16)hv;
        if (!writeY) hfinal[(size_t)m * H2 + ycol + n] = (__bf16)hv; // last = h_n
      }
    }
    __syncthreads();

    // y[b, t, ycol:ycol+256] <- hnxt : 64 rows x 256 bf16, row stride T*2H.
    if (writeY && wave == 0)
      tdm_store_2d((u32)(size_t)(void*)hnxt, y + (size_t)t * H2 + ycol,
                   (u32)HH, (u32)BB, (u32)TT * H2);
  }
}

// ---------------------------------------------------------------------------
extern "C" void kernel_launch(void* const* d_in, const int* in_sizes, int n_in,
                              void* d_out, int out_size, void* d_ws, size_t ws_size,
                              hipStream_t stream) {
  (void)in_sizes; (void)n_in; (void)out_size; (void)ws_size;

  const int*   text = (const int*)d_in[0];
  const float* emb  = (const float*)d_in[1];
  // dir order: 0=l0f, 1=l0b, 2=l1f, 3=l1b
  const float* Wih[4] = {(const float*)d_in[2],  (const float*)d_in[6],
                         (const float*)d_in[10], (const float*)d_in[14]};
  const float* Whh[4] = {(const float*)d_in[3],  (const float*)d_in[7],
                         (const float*)d_in[11], (const float*)d_in[15]};
  const float* bih[4] = {(const float*)d_in[4],  (const float*)d_in[8],
                         (const float*)d_in[12], (const float*)d_in[16]};
  const float* bhh[4] = {(const float*)d_in[5],  (const float*)d_in[9],
                         (const float*)d_in[13], (const float*)d_in[17]};
  const float* fcW = (const float*)d_in[18];
  const float* fcb = (const float*)d_in[19];
  float* out = (float*)d_out;

  char* ws = (char*)d_ws;
  size_t off = 0;
  auto alloc = [&](size_t bytes) -> char* {
    char* p = ws + off;
    off += (bytes + 255) & ~(size_t)255;
    return p;
  };

  __bf16* xb = (__bf16*)alloc((size_t)BT * EP * 2);            // 20 MB
  __bf16* Wih_b[4];
  Wih_b[0] = (__bf16*)alloc((size_t)G4 * EP * 2);
  Wih_b[1] = (__bf16*)alloc((size_t)G4 * EP * 2);
  Wih_b[2] = (__bf16*)alloc((size_t)G4 * H2 * 2);
  Wih_b[3] = (__bf16*)alloc((size_t)G4 * H2 * 2);
  __bf16* Whh_b[4];
  for (int d = 0; d < 4; ++d) Whh_b[d] = (__bf16*)alloc((size_t)G4 * HH * 2);
  __bf16* Wfc_b = (__bf16*)alloc((size_t)OO * H2 * 2);
  float* xgF = (float*)alloc((size_t)BT * G4 * 4);             // 128 MB (reused L0/L1)
  float* xgB = (float*)alloc((size_t)BT * G4 * 4);             // 128 MB
  __bf16* y0  = (__bf16*)alloc((size_t)BT * H2 * 2);           // 32 MB
  __bf16* hid = (__bf16*)alloc((size_t)BB * H2 * 2);

  // 1) embedding gather + bf16 cast (K padded to 320)
  embed_kernel<<<dim3(BT), dim3(EP), 0, stream>>>(text, emb, xb);

  // 2) weight casts
  auto cast = [&](const float* s, __bf16* d, int r, int c, int cp) {
    int nel = r * cp;
    castpad_kernel<<<dim3((nel + 255) / 256), dim3(256), 0, stream>>>(s, d, r, c, cp);
  };
  cast(Wih[0], Wih_b[0], G4, EE, EP);
  cast(Wih[1], Wih_b[1], G4, EE, EP);
  cast(Wih[2], Wih_b[2], G4, H2, H2);
  cast(Wih[3], Wih_b[3], G4, H2, H2);
  for (int d = 0; d < 4; ++d) cast(Whh[d], Whh_b[d], G4, HH, HH);
  cast(fcW, Wfc_b, OO, H2, H2);

  // 3) layer-0 input projections: xg = x @ Wih^T + bih   [32768 x 1024]
  dim3 gx(G4 / 128, BT / 64);
  wmma_gemm_bias<<<gx, dim3(256), 0, stream>>>(xb, Wih_b[0], bih[0], xgF, G4, EP);
  wmma_gemm_bias<<<gx, dim3(256), 0, stream>>>(xb, Wih_b[1], bih[1], xgB, G4, EP);

  // 4) layer-0 recurrence (both directions concurrently), emits y0 [BT, 2H]
  lstm_kernel<<<dim3(2), dim3(1024), 0, stream>>>(xgF, xgB, Whh_b[0], Whh_b[1],
                                                  bhh[0], bhh[1], y0, nullptr, 1);

  // 5) layer-1 input projections (reuse xg buffers)
  wmma_gemm_bias<<<gx, dim3(256), 0, stream>>>(y0, Wih_b[2], bih[2], xgF, G4, H2);
  wmma_gemm_bias<<<gx, dim3(256), 0, stream>>>(y0, Wih_b[3], bih[3], xgB, G4, H2);

  // 6) layer-1 recurrence; only final hidden states kept -> hid [64, 2H] bf16
  lstm_kernel<<<dim3(2), dim3(1024), 0, stream>>>(xgF, xgB, Whh_b[2], Whh_b[3],
                                                  bhh[2], bhh[3], nullptr, hid, 0);

  // 7) FC: out[64,512] = hid @ fcW^T + fcb  (f32 output)
  dim3 gf(OO / 128, BB / 64);
  wmma_gemm_bias<<<gf, dim3(256), 0, stream>>>(hid, Wfc_b, fcb, out, OO, H2);
}
